// TriangleAttention_42786464202820
// MI455X (gfx1250) — compile-verified, matrix operand order checked by reading, not run on hardware
//
#include <hip/hip_runtime.h>
#include <hip/hip_fp16.h>

typedef __attribute__((ext_vector_type(16))) _Float16 v16h;
typedef __attribute__((ext_vector_type(8)))  _Float16 v8h;
typedef __attribute__((ext_vector_type(8)))  float    v8f;

#define WMMA16(A, B, C) \
  __builtin_amdgcn_wmma_f32_16x16x32_f16(false, (A), false, (B), (short)0, (C), false, false)

// Load a 16-bit WMMA A/B fragment: per-lane 8 contiguous halfs at p and 8 at p+16.
// Caller supplies p = base + row*stride + (lane>>4)*8.
__device__ __forceinline__ v16h ldfrag(const _Float16* p) {
  v8h lo = *(const v8h*)p;
  v8h hi = *(const v8h*)(p + 16);
  return __builtin_shufflevector(lo, hi, 0,1,2,3,4,5,6,7,8,9,10,11,12,13,14,15);
}

// ---------------------------------------------------------------------------
// Kernel 0: convert + transpose weights to f16.  wT[m][n][c] = W_m[c][n]
// m: 0=Wq 1=Wk 2=Wv 3=Wg 4=Wout.  grid(128,5), block(128)
// ---------------------------------------------------------------------------
__global__ void k_prep(const float* __restrict__ Wq, const float* __restrict__ Wk,
                       const float* __restrict__ Wv, const float* __restrict__ Wg,
                       const float* __restrict__ Wout, _Float16* __restrict__ wT) {
  int n = blockIdx.x, m = blockIdx.y, c = threadIdx.x;
  const float* W = (m == 0) ? Wq : (m == 1) ? Wk : (m == 2) ? Wv : (m == 3) ? Wg : Wout;
  wT[((size_t)m * 128 + n) * 128 + c] = (_Float16)W[(size_t)c * 128 + n];
}

// ---------------------------------------------------------------------------
// Kernel 1: fused LayerNorm + Q/K/V/G projections (WMMA) + bias table.
// One block = 16 flattened (i,j) rows, 256 threads = 8 waves.
// ---------------------------------------------------------------------------
__global__ __launch_bounds__(256) void k_lnproj(
    const float* __restrict__ z, const int* __restrict__ mask,
    const float* __restrict__ gamma, const float* __restrict__ beta,
    const float* __restrict__ Wb, const _Float16* __restrict__ wT,
    _Float16* __restrict__ qh, _Float16* __restrict__ kh,
    _Float16* __restrict__ vT, float* __restrict__ sg, float* __restrict__ bm) {
  __shared__ alignas(16) float    znf[16][128];
  __shared__ alignas(16) _Float16 znh[16][136];   // padded row stride (272 B)

  const int tid = threadIdx.x, lane = tid & 31, w = tid >> 5;
  const int rowbase = blockIdx.x * 16;
  const int lrow = lane & 15, koff = (lane >> 4) * 8;

  // ---- LayerNorm: wave w handles rows 2w and 2w+1, 16 lanes x 8 floats each
  {
    const int r  = 2 * w + (lane >> 4);
    const int lc = lane & 15;
    const float* zr = z + (size_t)(rowbase + r) * 128 + lc * 8;
    float x[8], s = 0.f, ss = 0.f;
#pragma unroll
    for (int e = 0; e < 8; ++e) { x[e] = zr[e]; s += x[e]; ss += x[e] * x[e]; }
#pragma unroll
    for (int mlt = 8; mlt >= 1; mlt >>= 1) { s += __shfl_xor(s, mlt, 32); ss += __shfl_xor(ss, mlt, 32); }
    const float mu  = s * (1.f / 128.f);
    const float var = ss * (1.f / 128.f) - mu * mu;
    const float inv = rsqrtf(var + 1e-5f);
#pragma unroll
    for (int e = 0; e < 8; ++e) {
      const float y = (x[e] - mu) * inv * gamma[lc * 8 + e] + beta[lc * 8 + e];
      znf[r][lc * 8 + e] = y;
      znh[r][lc * 8 + e] = (_Float16)y;
    }
  }
  __syncthreads();

  // ---- bias table: bm[h][row] = zn . Wb[:,h] + NEG*(!mask[k]),  row=(j,k) pair
  if (tid < 64) {
    const int r = tid >> 2, h = tid & 3;
    float acc = 0.f;
#pragma unroll 8
    for (int c = 0; c < 128; ++c) acc += znf[r][c] * Wb[c * 4 + h];
    const int row = rowbase + r;
    const int kk  = row & 255;
    if (!mask[kk]) acc += -1e9f;
    bm[(size_t)h * 65536 + row] = acc;
  }

  // ---- projections: wave w -> matrix m = w>>1, 4 column tiles each
  const int m = w >> 1;
  const _Float16* WT = wT + (size_t)m * 16384;
#pragma unroll
  for (int q4 = 0; q4 < 4; ++q4) {
    const int ntile = (w & 1) * 4 + q4;
    const int ncol  = ntile * 16 + lrow;
    v8f acc = {};
#pragma unroll
    for (int ks = 0; ks < 4; ++ks) {
      v16h A = ldfrag(&znh[lrow][ks * 32 + koff]);
      v16h B = ldfrag(WT + (size_t)ncol * 128 + ks * 32 + koff);
      acc = WMMA16(A, B, acc);
    }
    const int h = ncol >> 5, c = ncol & 31;
#pragma unroll
    for (int r = 0; r < 8; ++r) {
      const int row = rowbase + r + 8 * (lane >> 4);
      const int i = row >> 8, j = row & 255;
      const size_t ihj = (size_t)(i * 4 + h) * 256 + j;
      const float val = acc[r];
      if      (m == 0) qh[ihj * 32 + c] = (_Float16)val;
      else if (m == 1) kh[ihj * 32 + c] = (_Float16)val;
      else if (m == 2) vT[((size_t)(i * 4 + h) * 32 + c) * 256 + j] = (_Float16)val;
      else             sg[ihj * 32 + c] = 1.f / (1.f + __expf(-val));
    }
  }
}

// ---------------------------------------------------------------------------
// Kernel 2: attention.  One wave = 16 query rows of one (i,h) slice.
// block = 8 waves, grid = (i*4+h)*2 + jhalf  => 2048 blocks.
// ---------------------------------------------------------------------------
__global__ __launch_bounds__(256) void k_attn(
    const _Float16* __restrict__ qh, const _Float16* __restrict__ kh,
    const _Float16* __restrict__ vTp, const float* __restrict__ sg,
    const float* __restrict__ bm, _Float16* __restrict__ og) {
  __shared__ alignas(16) _Float16 pbuf[8][16][264];   // per-wave P tile (528 B rows)

  const int tid = threadIdx.x, lane = tid & 31, w = tid >> 5;
  const int bid = blockIdx.x;
  const int jhalf = bid & 1, ih = bid >> 1;           // ih = i*4 + h
  const int jt = jhalf * 8 + w;                       // j-tile 0..15
  const int lrow = lane & 15, koff = (lane >> 4) * 8;
  const int h = ih & 3, i = ih >> 2;

  const _Float16* Qb = qh  + (size_t)ih * 256 * 32;
  const _Float16* Kb = kh  + (size_t)ih * 256 * 32;
  const _Float16* Vb = vTp + (size_t)ih * 32 * 256;
  const float*   bmh = bm  + (size_t)h * 65536;

  // ---- S = Q Kᵀ : full 16x256 strip in registers (K dim = C = 32, one WMMA/tile)
  const v16h A = ldfrag(Qb + (size_t)(jt * 16 + lrow) * 32 + koff);
  v8f s[16];
#pragma unroll
  for (int kt = 0; kt < 16; ++kt) {
    v16h B = ldfrag(Kb + (size_t)(kt * 16 + lrow) * 32 + koff);
    v8f zero = {};
    s[kt] = WMMA16(A, B, zero);
  }

  // ---- scale + bias + softmax over k (row = jt*16 + r + 8*(lane>>4))
  const float scale = 0.17677669529663687f;   // 1/sqrt(32)
#pragma unroll
  for (int r = 0; r < 8; ++r) {
    const int jrow = jt * 16 + r + 8 * (lane >> 4);
    const float* brow = bmh + (size_t)jrow * 256 + lrow;
    float mx = -3.0e38f;
#pragma unroll
    for (int kt = 0; kt < 16; ++kt) {
      s[kt][r] = s[kt][r] * scale + brow[kt * 16];
      mx = fmaxf(mx, s[kt][r]);
    }
#pragma unroll
    for (int mlt = 8; mlt >= 1; mlt >>= 1) mx = fmaxf(mx, __shfl_xor(mx, mlt, 32));
    float sum = 0.f;
#pragma unroll
    for (int kt = 0; kt < 16; ++kt) {
      const float e = __expf(s[kt][r] - mx);
      s[kt][r] = e;
      sum += e;
    }
#pragma unroll
    for (int mlt = 8; mlt >= 1; mlt >>= 1) sum += __shfl_xor(sum, mlt, 32);
    const float inv = 1.f / sum;
    const int prow = r + 8 * (lane >> 4);
#pragma unroll
    for (int kt = 0; kt < 16; ++kt)
      pbuf[w][prow][kt * 16 + lrow] = (_Float16)(s[kt][r] * inv);
  }
  __syncthreads();

  // ---- O = P V : 8 k-steps x 2 channel tiles
  v8f o0 = {}, o1 = {};
#pragma unroll
  for (int st = 0; st < 8; ++st) {
    v16h Ap = ldfrag(&pbuf[w][lrow][st * 32 + koff]);
    v16h B0 = ldfrag(Vb + (size_t)(lrow)      * 256 + st * 32 + koff);
    v16h B1 = ldfrag(Vb + (size_t)(16 + lrow) * 256 + st * 32 + koff);
    o0 = WMMA16(Ap, B0, o0);
    o1 = WMMA16(Ap, B1, o1);
  }

  // ---- gate by sigmoid(g), store f16 gated output og[(i*256+j)][h*32+c]
#pragma unroll
  for (int r = 0; r < 8; ++r) {
    const int jrow = jt * 16 + r + 8 * (lane >> 4);
    const size_t sb = ((size_t)ih * 256 + jrow) * 32;
    const size_t ob = ((size_t)i * 256 + jrow) * 128 + h * 32;
    og[ob + lrow]      = (_Float16)(o0[r] * sg[sb + lrow]);
    og[ob + 16 + lrow] = (_Float16)(o1[r] * sg[sb + 16 + lrow]);
  }
}

// ---------------------------------------------------------------------------
// Kernel 3: out = og[65536x128] @ Wout  (f16 WMMA, f32 out).  512 blocks x 8 waves.
// ---------------------------------------------------------------------------
__global__ __launch_bounds__(256) void k_out(const _Float16* __restrict__ og,
                                             const _Float16* __restrict__ WoT,
                                             float* __restrict__ out) {
  const int tid = threadIdx.x, lane = tid & 31, w = tid >> 5;
  const int rowbase = (blockIdx.x * 8 + w) * 16;
  const int lrow = lane & 15, koff = (lane >> 4) * 8;

  v8f acc[8] = {};
#pragma unroll
  for (int ks = 0; ks < 4; ++ks) {
    v16h A = ldfrag(og + (size_t)(rowbase + lrow) * 128 + ks * 32 + koff);
#pragma unroll
    for (int nt = 0; nt < 8; ++nt) {
      v16h B = ldfrag(WoT + (size_t)(nt * 16 + lrow) * 128 + ks * 32 + koff);
      acc[nt] = WMMA16(A, B, acc[nt]);
    }
  }
#pragma unroll
  for (int nt = 0; nt < 8; ++nt) {
    const int n = nt * 16 + lrow;
#pragma unroll
    for (int r = 0; r < 8; ++r) {
      const int row = rowbase + r + 8 * (lane >> 4);
      out[(size_t)row * 128 + n] = acc[nt][r];
    }
  }
}

// ---------------------------------------------------------------------------
extern "C" void kernel_launch(void* const* d_in, const int* in_sizes, int n_in,
                              void* d_out, int out_size, void* d_ws, size_t ws_size,
                              hipStream_t stream) {
  const float* z     = (const float*)d_in[0];
  const int*   mask  = (const int*)d_in[1];
  const float* gamma = (const float*)d_in[2];
  const float* beta  = (const float*)d_in[3];
  const float* Wq    = (const float*)d_in[4];
  const float* Wk    = (const float*)d_in[5];
  const float* Wv    = (const float*)d_in[6];
  const float* Wb    = (const float*)d_in[7];
  const float* Wg    = (const float*)d_in[8];
  const float* Wout  = (const float*)d_in[9];
  float* out = (float*)d_out;

  char* ws = (char*)d_ws;
  size_t off = 0;
  auto alloc = [&](size_t bytes) -> void* {
    void* p = ws + off;
    off = (off + bytes + 255) & ~(size_t)255;
    return p;
  };
  const size_t NHC = (size_t)4 * 256 * 256 * 32;      // 8,388,608 elems
  _Float16* wT = (_Float16*)alloc(5 * 16384 * sizeof(_Float16));
  _Float16* qh = (_Float16*)alloc(NHC * sizeof(_Float16));
  _Float16* kh = (_Float16*)alloc(NHC * sizeof(_Float16));
  _Float16* vT = (_Float16*)alloc(NHC * sizeof(_Float16));
  float*    sg = (float*)   alloc(NHC * sizeof(float));
  float*    bmp= (float*)   alloc((size_t)4 * 65536 * sizeof(float));
  _Float16* og = (_Float16*)alloc((size_t)65536 * 128 * sizeof(_Float16));

  k_prep  <<<dim3(128, 5), 128, 0, stream>>>(Wq, Wk, Wv, Wg, Wout, wT);
  k_lnproj<<<4096, 256, 0, stream>>>(z, mask, gamma, beta, Wb, wT, qh, kh, vT, sg, bmp);
  k_attn  <<<2048, 256, 0, stream>>>(qh, kh, vT, sg, bmp, og);
  k_out   <<<512,  256, 0, stream>>>(og, wT + (size_t)4 * 16384, out);
}
